// ChannelMamba_9363028706031
// MI455X (gfx1250) — matrix operand order repaired and checked by reading, not compiled, not run on hardware
//
#include <hip/hip_runtime.h>

#define B_  4
#define C_  192
#define L_  1024
#define NST 16
#define DTR 12
#define E2  384   // 2 * d_inner rows of xz
#define EP  96    // padded x_dbl rows: [0..43]=fwd proj, [48..91]=bwd proj, rest zero

typedef __attribute__((ext_vector_type(16))) _Float16 v16h;
typedef __attribute__((ext_vector_type(8)))  float    v8f;

// ---------------- workspace layout (in floats) ----------------
static const size_t N_XZ  = (size_t)B_ * E2 * L_;   // 1,572,864
static const size_t N_XC  = (size_t)B_ * C_ * L_;   //   786,432
static const size_t N_XD  = (size_t)B_ * L_ * EP;   //   393,216
static const size_t OFF_XZ = 0;
static const size_t OFF_XC = OFF_XZ + N_XZ;
static const size_t OFF_SZ = OFF_XC + N_XC;
static const size_t OFF_XD = OFF_SZ + N_XC;
static const size_t OFF_DL = OFF_XD + N_XD;         // delta, 2 directions
static const size_t OFF_OS = OFF_DL + 2 * N_XC;     // scan output, 2 directions
// total = OFF_OS + 2*N_XC = 6,684,672 floats ~= 25.5 MB

// ============================================================
// K1: xz[b,e,l] = sum_d W_in[e,d] * u[b,d,l]   (WMMA f16->f32)
//   per-wave 16x16 output tile, fully unrolled K: 6 WMMAs
// ============================================================
__global__ void k_xz_gemm(const float* __restrict__ u,
                          const float* __restrict__ Win,
                          float* __restrict__ xz) {
  const int lane = threadIdx.x & 31;
  const int wave = threadIdx.x >> 5;
  const int tile = blockIdx.x * 8 + wave;         // 6144 tiles total
  const int b  = tile / (24 * 64);
  const int r  = tile % (24 * 64);
  const int m0 = (r / 64) * 16;
  const int n0 = (r % 64) * 16;
  const int half = lane >> 4;
  const int ln   = lane & 15;

  const float* U    = u + (size_t)b * C_ * L_;    // U[d][l], ld = L_
  const float* arow = Win + (size_t)(m0 + ln) * C_;

  v8f acc = {};
#pragma unroll
  for (int k0 = 0; k0 < C_; k0 += 32) {
    v16h a, bm;
#pragma unroll
    for (int v = 0; v < 8; ++v) {
      const int ka = k0 + half * 16 + 2 * v;
      a[2 * v]     = (_Float16)arow[ka];
      a[2 * v + 1] = (_Float16)arow[ka + 1];
      const int kb = k0 + half * 16 + 2 * v;
      bm[2 * v]     = (_Float16)U[(size_t)kb * L_ + n0 + ln];
      bm[2 * v + 1] = (_Float16)U[(size_t)(kb + 1) * L_ + n0 + ln];
    }
    acc = __builtin_amdgcn_wmma_f32_16x16x32_f16(false, a, false, bm,
                                                 (short)0, acc, false, false);
  }
#pragma unroll
  for (int v = 0; v < 8; ++v) {
    const int e = m0 + v + 8 * half;
    xz[((size_t)b * E2 + e) * L_ + n0 + ln] = acc[v];
  }
}

// ============================================================
// K2: 3-tap depthwise conv along d (the 3x3 over a 1-high image
//     reduces to its middle row) + bias; also silu(z)
// ============================================================
__global__ void k_conv_silu(const float* __restrict__ xz,
                            const float* __restrict__ cw,
                            const float* __restrict__ cb,
                            float* __restrict__ xc,
                            float* __restrict__ szb) {
  const int idx = blockIdx.x * blockDim.x + threadIdx.x;  // over B_*C_*L_
  const int l = idx % L_;
  const int d = (idx / L_) % C_;
  const int b = idx / (L_ * C_);
  const float* xrow = xz + (size_t)b * E2 * L_;
  float acc = cb[l];
  const float w0 = cw[l * 9 + 3], w1 = cw[l * 9 + 4], w2 = cw[l * 9 + 5];
  if (d > 0)      acc += w0 * xrow[(size_t)(d - 1) * L_ + l];
  acc += w1 * xrow[(size_t)d * L_ + l];
  if (d < C_ - 1) acc += w2 * xrow[(size_t)(d + 1) * L_ + l];
  xc[idx] = acc;
  const float z = xrow[(size_t)(C_ + d) * L_ + l];
  szb[idx] = z / (1.f + __expf(-z));
}

// ============================================================
// K3: x_dbl[b,l,e] = sum_d Wcat[e,d] * xconv[b,d,l]  (WMMA)
//   Wcat rows: [0..43]=W_xproj, [48..91]=W_xproj_b, else 0
//   stored token-major so scan reads B/C contiguously
// ============================================================
__global__ void k_xdbl_gemm(const float* __restrict__ xc,
                            const float* __restrict__ Wx,
                            const float* __restrict__ Wxb,
                            float* __restrict__ xd) {
  const int lane = threadIdx.x & 31;
  const int wave = threadIdx.x >> 5;
  const int tile = blockIdx.x * 8 + wave;         // 1536 tiles total
  const int b  = tile / (6 * 64);
  const int r  = tile % (6 * 64);
  const int m0 = (r / 64) * 16;
  const int n0 = (r % 64) * 16;
  const int half = lane >> 4;
  const int ln   = lane & 15;

  const int e = m0 + ln;
  const float* arow = (e < 44) ? (Wx + (size_t)e * C_)
                   : (e >= 48 && e < 92) ? (Wxb + (size_t)(e - 48) * C_)
                   : nullptr;
  const float* X = xc + (size_t)b * C_ * L_;

  v8f acc = {};
#pragma unroll
  for (int k0 = 0; k0 < C_; k0 += 32) {
    v16h a, bm;
#pragma unroll
    for (int v = 0; v < 8; ++v) {
      const int ka = k0 + half * 16 + 2 * v;
      a[2 * v]     = (_Float16)(arow ? arow[ka]     : 0.f);
      a[2 * v + 1] = (_Float16)(arow ? arow[ka + 1] : 0.f);
      const int kb = k0 + half * 16 + 2 * v;
      bm[2 * v]     = (_Float16)X[(size_t)kb * L_ + n0 + ln];
      bm[2 * v + 1] = (_Float16)X[(size_t)(kb + 1) * L_ + n0 + ln];
    }
    acc = __builtin_amdgcn_wmma_f32_16x16x32_f16(false, a, false, bm,
                                                 (short)0, acc, false, false);
  }
#pragma unroll
  for (int v = 0; v < 8; ++v) {
    const int eo = m0 + v + 8 * half;
    const int l  = n0 + ln;
    xd[((size_t)b * L_ + l) * EP + eo] = acc[v];
  }
}

// ============================================================
// K4: delta[dir,b,d,l] = softplus( Wdt[d,:] . x_dbl[b,l,base:base+12]
//                                 + dt_bias[d] )
// ============================================================
__global__ void k_delta(const float* __restrict__ xd,
                        const float* __restrict__ Wdt,
                        const float* __restrict__ Wdtb,
                        const float* __restrict__ dtb,
                        float* __restrict__ delta) {
  const int idx = blockIdx.x * blockDim.x + threadIdx.x;  // over 2*B_*C_*L_
  const int l   = idx % L_;
  const int d   = (idx / L_) % C_;
  const int b   = (idx / (L_ * C_)) % B_;
  const int dir = idx / (L_ * C_ * B_);
  const float* w    = (dir ? Wdtb : Wdt) + (size_t)d * DTR;
  const float* xrow = xd + ((size_t)b * L_ + l) * EP + dir * 48;
  float acc = dtb[d];
#pragma unroll
  for (int r = 0; r < DTR; ++r) acc += w[r] * xrow[r];
  delta[idx] = (acc > 20.f) ? acc : log1pf(__expf(acc));
}

// ============================================================
// K5: selective scan. Half-wave = one (b,d,dir) row; lane n holds
// state n. 1024 steps in chunks of 16: lane n preloads the chunk's
// scalars for step l0+n (coalesced), broadcast per-step via shfl.
// The 16 exp()s per chunk are independent of h -> off critical path.
// Note: only x is reversed for dir==1 (matches reference).
// ============================================================
__global__ void k_scan(const float* __restrict__ xc,
                       const float* __restrict__ xd,
                       const float* __restrict__ delta,
                       const float* __restrict__ szb,
                       const float* __restrict__ Alog,
                       const float* __restrict__ Ablog,
                       const float* __restrict__ Dp,
                       const float* __restrict__ Dbp,
                       float* __restrict__ oscan) {
  const int lane = threadIdx.x & 31;
  const int wave = threadIdx.x >> 5;
  const int half = lane >> 4;
  const int n    = lane & 15;
  const int dir  = blockIdx.y;
  const int row  = blockIdx.x * 16 + wave * 2 + half;   // 0..767 = b*C_+d
  const int b = row / C_;
  const int d = row % C_;

  const float a  = -__expf((dir ? Ablog : Alog)[d * NST + n]);
  const float Dd = (dir ? Dbp : Dp)[d];
  const float* xrow  = xc  + (size_t)row * L_;
  const float* dlrow = delta + ((size_t)dir * B_ * C_ + row) * L_;
  const float* szrow = szb + (size_t)row * L_;
  const float* xdb   = xd + (size_t)b * L_ * EP + dir * 48;
  float* orow = oscan + ((size_t)dir * B_ * C_ + row) * L_;

  float h = 0.f;
  for (int l0 = 0; l0 < L_; l0 += 16) {
    // coalesced per-chunk scalar preload: lane n owns step l0+n
    const int lm = l0 + n;
    const float xv_m = xrow[dir ? (L_ - 1 - lm) : lm];
    const float dl_m = dlrow[lm];
    const float sz_m = szrow[lm];
    if (l0 + 16 < L_) {  // prefetch next chunk's B/C block
      __builtin_prefetch(&xdb[(size_t)(l0 + 16 + n) * EP + 12], 0, 1);
    }
    float oacc = 0.f;
#pragma unroll
    for (int s = 0; s < 16; ++s) {
      const float xv = __shfl(xv_m, s, 16);
      const float dl = __shfl(dl_m, s, 16);
      const float Bv = xdb[(size_t)(l0 + s) * EP + 12 + n];
      const float Cv = xdb[(size_t)(l0 + s) * EP + 28 + n];
      h = h * __expf(dl * a) + dl * Bv * xv;   // the only serial dependence
      float y = h * Cv;
      y += __shfl_xor(y, 1, 16);
      y += __shfl_xor(y, 2, 16);
      y += __shfl_xor(y, 4, 16);
      y += __shfl_xor(y, 8, 16);
      if (n == s) oacc = (y + xv * Dd) * sz_m; // lane s keeps step l0+s
    }
    orow[l0 + n] = oacc;                       // coalesced 64B store per chunk
  }
}

// ============================================================
// K6: per-(b,d) layernorm over l for both directions, gate with
// silu(z), combine fwd + reversed-bwd.
// ============================================================
__global__ void k_final(const float* __restrict__ oscan,
                        const float* __restrict__ szb,
                        const float* __restrict__ lnw,  const float* __restrict__ lnb,
                        const float* __restrict__ ln1w, const float* __restrict__ ln1b,
                        float* __restrict__ out) {
  const int row = blockIdx.x;                 // b*C_+d
  const int tid = threadIdx.x;                // 256 threads
  const float* of  = oscan + (size_t)row * L_;
  const float* ob  = oscan + (size_t)(B_ * C_) * L_ + (size_t)row * L_;
  const float* szr = szb + (size_t)row * L_;

  float sf = 0.f, qf = 0.f, sb = 0.f, qb = 0.f;
  for (int l = tid; l < L_; l += 256) {
    const float vf = of[l], vb = ob[l];
    sf += vf; qf += vf * vf; sb += vb; qb += vb * vb;
  }
  for (int off = 16; off > 0; off >>= 1) {
    sf += __shfl_xor(sf, off, 32);
    qf += __shfl_xor(qf, off, 32);
    sb += __shfl_xor(sb, off, 32);
    qb += __shfl_xor(qb, off, 32);
  }
  __shared__ float red[4][8];
  const int wv = tid >> 5, lane = tid & 31;
  if (lane == 0) { red[0][wv] = sf; red[1][wv] = qf;
                   red[2][wv] = sb; red[3][wv] = qb; }
  __syncthreads();
  sf = qf = sb = qb = 0.f;
#pragma unroll
  for (int i = 0; i < 8; ++i) { sf += red[0][i]; qf += red[1][i];
                                sb += red[2][i]; qb += red[3][i]; }
  const float inv = 1.f / (float)L_;
  const float mf = sf * inv, mb = sb * inv;
  const float rf = rsqrtf(qf * inv - mf * mf + 1e-5f);
  const float rb = rsqrtf(qb * inv - mb * mb + 1e-5f);

  for (int l = tid; l < L_; l += 256) {
    const float t1 = ((of[l] - mf) * rf * lnw[l] + lnb[l]) * szr[l];
    const int lr = L_ - 1 - l;
    const float t2 = ((ob[lr] - mb) * rb * ln1w[lr] + ln1b[lr]) * szr[lr];
    out[(size_t)row * L_ + l] = t1 + t2;
  }
}

// ============================================================
extern "C" void kernel_launch(void* const* d_in, const int* in_sizes, int n_in,
                              void* d_out, int out_size, void* d_ws, size_t ws_size,
                              hipStream_t stream) {
  (void)in_sizes; (void)n_in; (void)out_size; (void)ws_size;
  const float* u     = (const float*)d_in[0];
  const float* Win   = (const float*)d_in[1];
  const float* cw    = (const float*)d_in[2];
  const float* cb    = (const float*)d_in[3];
  const float* Wx    = (const float*)d_in[4];
  const float* Wdt   = (const float*)d_in[5];
  const float* dtb   = (const float*)d_in[6];
  const float* Alog  = (const float*)d_in[7];
  const float* Dp    = (const float*)d_in[8];
  const float* Wxb   = (const float*)d_in[9];
  const float* Wdtb  = (const float*)d_in[10];
  const float* Ablog = (const float*)d_in[11];
  const float* Dbp   = (const float*)d_in[12];
  const float* lnw   = (const float*)d_in[13];
  const float* lnb   = (const float*)d_in[14];
  const float* ln1w  = (const float*)d_in[15];
  const float* ln1b  = (const float*)d_in[16];

  float* ws  = (float*)d_ws;
  float* xz  = ws + OFF_XZ;
  float* xc  = ws + OFF_XC;
  float* szb = ws + OFF_SZ;
  float* xd  = ws + OFF_XD;
  float* dl  = ws + OFF_DL;
  float* os  = ws + OFF_OS;
  float* out = (float*)d_out;

  k_xz_gemm  <<<768, 256, 0, stream>>>(u, Win, xz);
  k_conv_silu<<<3072, 256, 0, stream>>>(xz, cw, cb, xc, szb);
  k_xdbl_gemm<<<192, 256, 0, stream>>>(xc, Wx, Wxb, xd);
  k_delta    <<<6144, 256, 0, stream>>>(xd, Wdt, Wdtb, dtb, dl);
  dim3 g5(48, 2);
  k_scan     <<<g5, 256, 0, stream>>>(xc, xd, dl, szb, Alog, Ablog, Dp, Dbp, os);
  k_final    <<<768, 256, 0, stream>>>(os, szb, lnw, lnb, ln1w, ln1b, out);
}
